// BiGCNN_47347719471740
// MI455X (gfx1250) — matrix operand order repaired and checked by reading, not compiled, not run on hardware
//
#include <hip/hip_runtime.h>

#define NUv 100000
#define NVc 50000
#define NEe 500000
#define BM 64

typedef __attribute__((ext_vector_type(16))) _Float16 v16h;
typedef __attribute__((ext_vector_type(8)))  float    v8f;
typedef __attribute__((ext_vector_type(4)))  _Float16 h4;
typedef unsigned int u32x4 __attribute__((ext_vector_type(4)));
typedef unsigned int u32x8 __attribute__((ext_vector_type(8)));

__device__ __forceinline__ float wred(float v) {
#pragma unroll
  for (int m = 16; m > 0; m >>= 1) v += __shfl_xor(v, m, 32);
  return v;
}

// Tensor Data Mover: global -> LDS tile copy (CDNA5 ISA ch.8, D# groups 0/1).
// data_size = 8 bytes; dim0/tile_d0/stride in 8-byte units. 2D only (groups 2/3 NULL).
// Rows beyond `dim1` read as zero (hardware OOB padding).
__device__ __forceinline__ void tdm_load(unsigned lds_off, const void* gsrc,
                                         unsigned dim0_u8, unsigned tile_d0,
                                         unsigned dim1, unsigned tile_d1,
                                         unsigned stride_u8) {
  unsigned long long ga = (unsigned long long)(size_t)gsrc;
  u32x4 g0;
  g0[0] = 1u;                                            // count=1, user descriptor
  g0[1] = lds_off;                                       // lds_addr [63:32]
  g0[2] = (unsigned)ga;                                  // global_addr [95:64]
  g0[3] = ((unsigned)(ga >> 32) & 0x01FFFFFFu)           // global_addr [120:96]
          | 0x80000000u;                                 // type=2 ("image") [127:126]
  u32x8 g1;
  g1[0] = 0x00030000u;                                   // workgroup_mask=0, data_size=3 (8B)
  g1[1] = (dim0_u8 & 0xFFFFu) << 16;                     // tensor_dim0 [79:48] lo
  g1[2] = (dim0_u8 >> 16) | ((dim1 & 0xFFFFu) << 16);    // dim0 hi | tensor_dim1 lo
  g1[3] = (dim1 >> 16) | (tile_d0 << 16);                // dim1 hi | tile_dim0 [127:112]
  g1[4] = tile_d1 & 0xFFFFu;                             // tile_dim1 [143:128], tile_dim2=0
  g1[5] = stride_u8;                                     // tensor_dim0_stride lo32
  g1[6] = 0u;                                            // stride hi | dim1_stride lo
  g1[7] = 0u;
  asm volatile("tensor_load_to_lds %0, %1" :: "s"(g0), "s"(g1) : "memory");
}

// ---------------- elementwise helpers ----------------

__global__ __launch_bounds__(256) void cast_f16_kernel(const float* __restrict__ x,
                                                       _Float16* __restrict__ y, long n4) {
  long i = (long)blockIdx.x * 256 + threadIdx.x;
  if (i >= n4) return;
  float4 v = ((const float4*)x)[i];
  h4 o = { (_Float16)v.x, (_Float16)v.y, (_Float16)v.z, (_Float16)v.w };
  ((h4*)y)[i] = o;
}

__global__ __launch_bounds__(256) void zero_kernel(float* __restrict__ y, long n4) {
  long i = (long)blockIdx.x * 256 + threadIdx.x;
  if (i >= n4) return;
  ((float4*)y)[i] = make_float4(0.f, 0.f, 0.f, 0.f);
}

// One wave32 per 128-wide row: LayerNorm -> f16 activation.
__global__ __launch_bounds__(256) void ln_f16_kernel(const float* __restrict__ x,
    const float* __restrict__ g, const float* __restrict__ b,
    _Float16* __restrict__ y, long rows) {
  int wave = threadIdx.x >> 5, lane = threadIdx.x & 31;
  long row = (long)blockIdx.x * 8 + wave;
  if (row >= rows) return;
  float4 xv = ((const float4*)(x + row * 128))[lane];
  float s = xv.x + xv.y + xv.z + xv.w;
  float q = xv.x*xv.x + xv.y*xv.y + xv.z*xv.z + xv.w*xv.w;
  s = wred(s); q = wred(q);
  float mu = s * (1.f / 128.f);
  float rstd = rsqrtf(q * (1.f / 128.f) - mu * mu + 1e-5f);
  float4 gv = ((const float4*)g)[lane];
  float4 bv = ((const float4*)b)[lane];
  h4 o = { (_Float16)((xv.x - mu) * rstd * gv.x + bv.x),
           (_Float16)((xv.y - mu) * rstd * gv.y + bv.y),
           (_Float16)((xv.z - mu) * rstd * gv.z + bv.z),
           (_Float16)((xv.w - mu) * rstd * gv.w + bv.w) };
  ((h4*)(y + row * 128))[lane] = o;
}

// j0 = v_t[e_u] + e_t + c_t[e_v]; relu; LN(g_join_ln) -> f16 activation.
__global__ __launch_bounds__(256) void gather_join_kernel(const float* __restrict__ vt,
    const float* __restrict__ et, const float* __restrict__ ct,
    const int* __restrict__ eu, const int* __restrict__ ev,
    const float* __restrict__ g, const float* __restrict__ b,
    _Float16* __restrict__ y, long ne) {
  int wave = threadIdx.x >> 5, lane = threadIdx.x & 31;
  long e = (long)blockIdx.x * 8 + wave;
  if (e >= ne) return;
  int u = eu[e];
  int v = ev[e];
  float4 a = ((const float4*)(vt + (long)u * 128))[lane];
  float4 ee = ((const float4*)(et + e * 128))[lane];
  float4 c = ((const float4*)(ct + (long)v * 128))[lane];
  float4 xv;
  xv.x = fmaxf(a.x + ee.x + c.x, 0.f);
  xv.y = fmaxf(a.y + ee.y + c.y, 0.f);
  xv.z = fmaxf(a.z + ee.z + c.z, 0.f);
  xv.w = fmaxf(a.w + ee.w + c.w, 0.f);
  float s = xv.x + xv.y + xv.z + xv.w;
  float q = xv.x*xv.x + xv.y*xv.y + xv.z*xv.z + xv.w*xv.w;
  s = wred(s); q = wred(q);
  float mu = s * (1.f / 128.f);
  float rstd = rsqrtf(q * (1.f / 128.f) - mu * mu + 1e-5f);
  float4 gv = ((const float4*)g)[lane];
  float4 bv = ((const float4*)b)[lane];
  h4 o = { (_Float16)((xv.x - mu) * rstd * gv.x + bv.x),
           (_Float16)((xv.y - mu) * rstd * gv.y + bv.y),
           (_Float16)((xv.z - mu) * rstd * gv.z + bv.z),
           (_Float16)((xv.w - mu) * rstd * gv.w + bv.w) };
  ((h4*)(y + e * 128))[lane] = o;
}

// agg[idx[e]] += j[e]   (segment sum; agg fits in the 192MB L2)
__global__ __launch_bounds__(256) void scatter_kernel(const float* __restrict__ j,
    const int* __restrict__ idx, float* __restrict__ agg, long ne) {
  int wave = threadIdx.x >> 5, lane = threadIdx.x & 31;
  long e = (long)blockIdx.x * 8 + wave;
  if (e >= ne) return;
  int d = idx[e];
  float4 v = ((const float4*)(j + e * 128))[lane];
  float* p = agg + (long)d * 128 + lane * 4;
  atomicAdd(p + 0, v.x);
  atomicAdd(p + 1, v.y);
  atomicAdd(p + 2, v.z);
  atomicAdd(p + 3, v.w);
}

// ---------------- WMMA GEMM: out[M,128] = A[M, KT*32] @ W + epilogue ----------------
// KT = K/32 (4 or 8).  K=256 takes concat input from (A0 | A1), staged as two
// separate 64x128 LDS blocks.  All staging is done by the Tensor Data Mover.
// EPI 0: out = acc (+bias)
// EPI 1: out = LN(acc + bias; g, bln)
// EPI 2: out = resid + LN(relu(acc + bias); g, bln)
template<int KT, int EPI>
__global__ __launch_bounds__(256) void gemm_kernel(
    const _Float16* __restrict__ A0, const _Float16* __restrict__ A1,
    const _Float16* __restrict__ W, const float* __restrict__ bias,
    const float* __restrict__ g, const float* __restrict__ bln,
    const float* __restrict__ resid, float* __restrict__ out, int M) {
  constexpr int K = KT * 32;
  extern __shared__ char smem[];
  _Float16* Asl = (_Float16*)smem;              // KT==4: 64x128; KT==8: 2x(64x128)
  _Float16* Wsl = Asl + BM * K;                 // K x 128 f16, row-major
  const int tid = threadIdx.x;
  const int wave = tid >> 5, lane = tid & 31;
  const int m16 = lane & 15, hh = lane >> 4;
  const long row0 = (long)blockIdx.x * BM;

  // --- TDM staging: wave 0 issues async tile DMAs, then publishes via barrier ---
  if (tid < 32) {
    const unsigned aoff = (unsigned)(size_t)(void*)Asl;   // LDS aperture: low 32 bits
    const unsigned woff = (unsigned)(size_t)(void*)Wsl;
    const unsigned rows_left = (unsigned)(M - (int)row0); // OOB rows -> zero fill
    // A tile: 64 rows x 256B (32 x 8B units per row)
    tdm_load(aoff, A0 + row0 * 128, 32u, 32u, rows_left, 64u, 32u);
    if constexpr (KT == 8)
      tdm_load(aoff + BM * 128 * 2, A1 + row0 * 128, 32u, 32u, rows_left, 64u, 32u);
    // Weights: contiguous 1-D block of K*32 x 8B units
    tdm_load(woff, W, (unsigned)(K * 32), (unsigned)(K * 32), 1u, 0u,
             (unsigned)(K * 32));
    __builtin_amdgcn_s_wait_tensorcnt(0);
  }
  __syncthreads();

  // 4 row-blocks x 8 col-blocks of 16x16 tiles; each wave owns 4 tiles.
  v8f acc[4];
#pragma unroll
  for (int i = 0; i < 4; ++i) {
    int t = wave + 8 * i;
    int rb = t >> 3, cb = t & 7;
    v8f c = {0.f, 0.f, 0.f, 0.f, 0.f, 0.f, 0.f, 0.f};
#pragma unroll
    for (int kc = 0; kc < KT; ++kc) {
      const _Float16* ablk = Asl + ((KT == 8 && kc >= 4) ? BM * 128 : 0);
      const int kcl = (KT == 8) ? (kc & 3) : kc;
      const _Float16* arow = ablk + (rb * 16 + m16) * 128;
      v16h a, b;
      // A frag (16-bit A 16x32 layout): lane<->M, half-lane picks K octet
      *(uint4*)&a       = *(const uint4*)(arow + kcl * 32 + hh * 8);
      *((uint4*)&a + 1) = *(const uint4*)(arow + kcl * 32 + 16 + hh * 8);
      // B frag: lane<->N, half-lane picks K octet (same K order as A), strided weight rows
      const _Float16* brow = Wsl + (kc * 32) * 128 + cb * 16 + m16;
#pragma unroll
      for (int s = 0; s < 16; ++s)
        b[s] = brow[((s < 8) ? (hh * 8 + s) : (16 + hh * 8 + (s - 8))) * 128];
      c = __builtin_amdgcn_wmma_f32_16x16x32_f16(false, a, false, b, (short)0, c,
                                                 false, false);
    }
    acc[i] = c;
  }

  if constexpr (EPI == 0) {
#pragma unroll
    for (int i = 0; i < 4; ++i) {
      int t = wave + 8 * i;
      int rb = t >> 3, cb = t & 7;
      int n = cb * 16 + m16;
      float bv = bias ? bias[n] : 0.f;
#pragma unroll
      for (int r = 0; r < 8; ++r) {
        long gm = row0 + rb * 16 + r + 8 * hh;
        if (gm < M) out[gm * 128 + n] = acc[i][r] + bv;
      }
    }
  } else {
    // LN epilogue: park the f32 tile in LDS (overlay on A/W stage), then per-row LN.
    __syncthreads();
    float* Csl = (float*)smem;                  // 64x128 f32 = 32KB
#pragma unroll
    for (int i = 0; i < 4; ++i) {
      int t = wave + 8 * i;
      int rb = t >> 3, cb = t & 7;
      int n = cb * 16 + m16;
      float bv = bias[n];
#pragma unroll
      for (int r = 0; r < 8; ++r) {
        float vv = acc[i][r] + bv;
        if (EPI == 2) vv = fmaxf(vv, 0.f);
        Csl[(rb * 16 + r + 8 * hh) * 128 + n] = vv;
      }
    }
    __syncthreads();
#pragma unroll
    for (int rr = 0; rr < 8; ++rr) {
      int row = wave * 8 + rr;
      float4 xv = ((const float4*)(Csl + row * 128))[lane];
      float s = xv.x + xv.y + xv.z + xv.w;
      float q = xv.x*xv.x + xv.y*xv.y + xv.z*xv.z + xv.w*xv.w;
      s = wred(s); q = wred(q);
      float mu = s * (1.f / 128.f);
      float rstd = rsqrtf(q * (1.f / 128.f) - mu * mu + 1e-5f);
      long gr = row0 + row;
      if (gr < M) {
        float4 gv = ((const float4*)g)[lane];
        float4 bv = ((const float4*)bln)[lane];
        float4 o;
        o.x = (xv.x - mu) * rstd * gv.x + bv.x;
        o.y = (xv.y - mu) * rstd * gv.y + bv.y;
        o.z = (xv.z - mu) * rstd * gv.z + bv.z;
        o.w = (xv.w - mu) * rstd * gv.w + bv.w;
        if constexpr (EPI == 2) {
          float4 hv = ((const float4*)(resid + gr * 128))[lane];
          o.x += hv.x; o.y += hv.y; o.z += hv.z; o.w += hv.w;
        }
        ((float4*)(out + gr * 128))[lane] = o;
      }
    }
  }
}

// ---------------- host launcher ----------------
//
// Workspace layout (bytes), total = 512,196,608:
//   [0,           128,000,000)  ACT   f16 act buffer (reused; merge: h@0, agg@64M)
//   [128,000,000, 179,200,000)  v_t   f32 [NU,128]
//   [179,200,000, 435,200,000)  e_t / j f32 [NE,128]
//   [435,200,000, 460,800,000)  c_t   f32 [NV,128]
//   [460,800,000, 512,000,000)  agg   f32 [<=NU,128]
//   [512,000,000, ...)          f16 weights
extern "C" void kernel_launch(void* const* d_in, const int* in_sizes, int n_in,
                              void* d_out, int out_size, void* d_ws, size_t ws_size,
                              hipStream_t stream) {
  (void)in_sizes; (void)n_in; (void)out_size; (void)ws_size;
  const float* v_emb  = (const float*)d_in[0];
  const float* e_emb  = (const float*)d_in[1];
  const float* c_emb  = (const float*)d_in[2];
  const int*   e_u    = (const int*)d_in[3];
  const int*   e_v    = (const int*)d_in[4];
  const float* W_left = (const float*)d_in[5];
  const float* b_left = (const float*)d_in[6];
  const float* W_edge = (const float*)d_in[7];
  const float* W_right= (const float*)d_in[8];
  const float* W_join = (const float*)d_in[9];
  const float* b_join = (const float*)d_in[10];
  const float* W_merge= (const float*)d_in[11];
  const float* b_merge= (const float*)d_in[12];
  const float* g_var  = (const float*)d_in[13];
  const float* b_var  = (const float*)d_in[14];
  const float* g_edge = (const float*)d_in[15];
  const float* b_edge = (const float*)d_in[16];
  const float* g_con  = (const float*)d_in[17];
  const float* b_con  = (const float*)d_in[18];
  const float* g_jl   = (const float*)d_in[19];
  const float* b_jl   = (const float*)d_in[20];
  const float* g_jt   = (const float*)d_in[21];
  const float* b_jt   = (const float*)d_in[22];
  const float* g_mg   = (const float*)d_in[23];
  const float* b_mgln = (const float*)d_in[24];

  char* ws = (char*)d_ws;
  _Float16* ACT  = (_Float16*)(ws);
  _Float16* ACTa = (_Float16*)(ws + 64000000);
  float* vt  = (float*)(ws + 128000000);
  float* et  = (float*)(ws + 179200000);
  float* ct  = (float*)(ws + 435200000);
  float* agg = (float*)(ws + 460800000);
  _Float16* Wl16 = (_Float16*)(ws + 512000000);
  _Float16* We16 = Wl16 + 16384;
  _Float16* Wr16 = Wl16 + 32768;
  _Float16* Wj16 = Wl16 + 49152;
  _Float16* Wm16 = Wl16 + 65536;

  const dim3 B(256);
  auto cvt = [&](const float* src, _Float16* dst, long n) {
    long n4 = n / 4;
    cast_f16_kernel<<<dim3((unsigned)((n4 + 255) / 256)), B, 0, stream>>>(src, dst, n4);
  };
  cvt(W_left, Wl16, 16384);
  cvt(W_edge, We16, 16384);
  cvt(W_right, Wr16, 16384);
  cvt(W_join, Wj16, 16384);
  cvt(W_merge, Wm16, 32768);

  float* out_var = (float*)d_out;
  float* out_con = out_var + (size_t)NUv * 128;

  const size_t SH4 = (size_t)BM * 128 * 2 + 128 * 128 * 2;   // 49152 B
  const size_t SH8 = (size_t)BM * 256 * 2 + 256 * 128 * 2;   // 98304 B

  for (int p = 0; p < 2; ++p) {
    const float* c_in  = (p == 0) ? c_emb : out_con;
    const int* idx_agg = (p == 0) ? e_v : e_u;
    long n_h           = (p == 0) ? NVc : NUv;
    float* h           = (p == 0) ? ct : vt;
    float* outp        = (p == 0) ? out_con : out_var;

    // v_t = LN(v) @ W_left + b_left
    ln_f16_kernel<<<dim3((NUv + 7) / 8), B, 0, stream>>>(v_emb, g_var, b_var, ACT, NUv);
    gemm_kernel<4, 0><<<dim3((NUv + BM - 1) / BM), B, SH4, stream>>>(
        ACT, nullptr, Wl16, b_left, nullptr, nullptr, nullptr, vt, NUv);
    // e_t = LN(e) @ W_edge
    ln_f16_kernel<<<dim3((NEe + 7) / 8), B, 0, stream>>>(e_emb, g_edge, b_edge, ACT, NEe);
    gemm_kernel<4, 0><<<dim3((NEe + BM - 1) / BM), B, SH4, stream>>>(
        ACT, nullptr, We16, nullptr, nullptr, nullptr, nullptr, et, NEe);
    // c_t = LN(c) @ W_right
    ln_f16_kernel<<<dim3((NVc + 7) / 8), B, 0, stream>>>(c_in, g_con, b_con, ACT, NVc);
    gemm_kernel<4, 0><<<dim3((NVc + BM - 1) / BM), B, SH4, stream>>>(
        ACT, nullptr, Wr16, nullptr, nullptr, nullptr, nullptr, ct, NVc);
    // jact = LN(relu(v_t[e_u] + e_t + c_t[e_v]))
    gather_join_kernel<<<dim3((NEe + 7) / 8), B, 0, stream>>>(
        vt, et, ct, e_u, e_v, g_jl, b_jl, ACT, NEe);
    // j = LN(jact @ W_join + b_join)   (written over e_t, which is consumed)
    gemm_kernel<4, 1><<<dim3((NEe + BM - 1) / BM), B, SH4, stream>>>(
        ACT, nullptr, Wj16, b_join, g_jt, b_jt, nullptr, et, NEe);
    // agg = segment_sum(j, idx_agg)
    zero_kernel<<<dim3((unsigned)((n_h * 32 + 255) / 256)), B, 0, stream>>>(agg, n_h * 32);
    scatter_kernel<<<dim3((NEe + 7) / 8), B, 0, stream>>>(et, idx_agg, agg, NEe);
    // out = h + LN(relu([h | agg] @ W_merge + b_merge))
    cvt(h, ACT, n_h * 128);
    cvt(agg, ACTa, n_h * 128);
    gemm_kernel<8, 2><<<dim3((unsigned)((n_h + BM - 1) / BM)), B, SH8, stream>>>(
        ACT, ACTa, Wm16, b_merge, g_mg, b_mgln, h, outp, (int)n_h);
  }
}
